// GRU_87582973100090
// MI455X (gfx1250) — matrix operand order repaired and checked by reading, not compiled, not run on hardware
//
#include <hip/hip_runtime.h>
#include <hip/hip_bf16.h>
#include <cstdint>
#include <cstddef>

// ---------------------------------------------------------------------------
// GRU on MI455X (gfx1250): bf16 WMMA GEMMs, fp32 state.
//   Phase 0: pack weights -> bf16 PRE-TRANSPOSED (Bt[N][K], fragment-ready),
//            x -> bf16, h0 -> fp32+bf16 state.
//   Phase 1 (parallel): Xzr = x@[Wzx|Wrx]+b, Xh = x@Whx+bh (M = T*B = 32768).
//   Phase 2 (serial, 512 x 2 launches): fused gate GEMM (sigmoid, h*r) and
//            fused update GEMM (tanh, h blend), epilogues inline.
//   Phase 3 (parallel): Y = H@Wout + bout -> d_out, copy h_final.
//
// GEMM core: 128 threads = 4 wave32; block tile 64x128; wave tile 64x32
// (4x2 accumulators => 8 v_wmma per K-chunk per wave; 8B/lane global B per
// WMMA). A staged in LDS, double-buffered, 1 barrier per 32-K chunk; B frags
// loaded directly global->VGPR from pre-transposed weights. Last K-chunk is
// peeled so the steady-state loop is branch-free.
// ---------------------------------------------------------------------------

typedef __bf16 bf16_t;
typedef __attribute__((ext_vector_type(16))) __bf16     v16bf;
typedef __attribute__((ext_vector_type(8)))  float       v8f;
typedef __attribute__((ext_vector_type(4)))  unsigned int u32x4;

#define T_STEPS 512
#define BATCH   64
#define IN_DIM  512
#define HID     1024
#define OUT_DIM 512

union Frag {
    v16bf  v;
    u32x4  q[2];
    bf16_t e[16];
};

__device__ __forceinline__ float sigmoidf_(float x) {
    return 1.0f / (1.0f + __expf(-x));
}
__device__ __forceinline__ float tanhf_(float x) {
    float e = __expf(2.0f * x);
    return (e - 1.0f) / (e + 1.0f);
}

// ---------------------------------------------------------------------------
// C[M,N] = A[M,K](bf16,row-major) * Bt[N,K](bf16, PRE-TRANSPOSED row-major)
// Requirements: M%64==0, N%128==0, K%32==0 (all call sites satisfy this).
// EPI: 0 = fp32 out (+bias)  -> Cf
//      1 = bf16 out (+bias)  -> Cb
//      2 = GRU gates: acc+=Xt; s=sigmoid; col<HID -> zbuf(f32);
//                     col>=HID -> aux = bf16(hbuf * s)   (h*r operand)
//      3 = GRU update: acc+=Xt; ht=tanh; hn=(1-z)*h+z*ht;
//                     hbuf=hn (f32), aux=bf16(hn), aux2=H_t
// ---------------------------------------------------------------------------
template<int EPI>
__global__ __launch_bounds__(128) void gemm_bf16_wmma(
    const bf16_t* __restrict__ A, const bf16_t* __restrict__ Bt,
    const float*  __restrict__ bias,
    float* __restrict__ Cf, bf16_t* __restrict__ Cb,
    const bf16_t* __restrict__ Xt,
    float* __restrict__ zbuf, float* __restrict__ hbuf,
    bf16_t* __restrict__ aux, bf16_t* __restrict__ aux2,
    int M, int N, int K)
{
    // Row stride 40 bf16 (80B): b128 LDS accesses stay 16B-aligned and rows
    // land 4 banks apart (conflict-free 16-lane row fan-out).
    __shared__ __align__(16) bf16_t ldsA[2][64 * 40];   // double-buffered [m][k]

    const int tid  = threadIdx.x;
    const int lane = tid & 31;
    const int wave = tid >> 5;           // 0..3
    const int wn   = wave * 32;          // wave N offset (wave tile 64x32)
    const int m0   = blockIdx.y * 64;
    const int n0   = blockIdx.x * 128;

    const int lr = lane & 15;            // 0..15
    const int hi = lane >> 4;            // 0 or 1 (half-wave)

    v8f acc[4][2];
#pragma unroll
    for (int r = 0; r < 4; ++r)
#pragma unroll
        for (int c = 0; c < 2; ++c)
#pragma unroll
            for (int i = 0; i < 8; ++i) acc[r][c][i] = 0.0f;

    // Cooperative A-tile load: 64x32 chunk, 128 threads -> 2 b128 each.
    const int arow = tid >> 1;           // 0..63
    const int aq   = (tid & 1) * 16;     // element offset 0 / 16 within chunk
    const bf16_t* Abase = A + (size_t)(m0 + arow) * K + aq;
    const int lA_off = arow * 40 + aq;

    // B fragment bases (contiguous k order, same as A): lane = column n = lr; lanes<16 K=0..15,
    // lanes>=16 K=16..31 -> 16 contiguous bf16 in a Bt row => two b128 loads.
    const bf16_t* Bfrag0 = Bt + (size_t)(n0 + wn + lr) * K + hi * 16;
    const bf16_t* Bfrag1 = Bfrag0 + (size_t)16 * K;

    const int nk = K >> 5;               // >= 16 at every call site

    // Preload A chunk 0
    *(u32x4*)&ldsA[0][lA_off]     = *(const u32x4*)(Abase);
    *(u32x4*)&ldsA[0][lA_off + 8] = *(const u32x4*)(Abase + 8);
    __syncthreads();

    Frag a[4], b[2];
    // Steady state: branch-free (last chunk peeled).
    for (int i = 0; i < nk - 1; ++i) {
        const int kc = i << 5;

        // Prefetch next A chunk into registers (no barrier dependence).
        u32x4 an0 = *(const u32x4*)(Abase + kc + 32);
        u32x4 an1 = *(const u32x4*)(Abase + kc + 40);

        // B fragments straight from global (L2-resident weights).
        b[0].q[0] = *(const u32x4*)(Bfrag0 + kc);
        b[0].q[1] = *(const u32x4*)(Bfrag0 + kc + 8);
        b[1].q[0] = *(const u32x4*)(Bfrag1 + kc);
        b[1].q[1] = *(const u32x4*)(Bfrag1 + kc + 8);

        // A fragments from LDS (contiguous k order, same as B): lane m=lr; lanes<16 hold
        // K=0..15, lanes>=16 hold K=16..31.
        const bf16_t* lA = &ldsA[i & 1][0];
#pragma unroll
        for (int r = 0; r < 4; ++r) {
            const bf16_t* pa = &lA[(r * 16 + lr) * 40 + hi * 16];
            a[r].q[0] = *(const u32x4*)(pa);
            a[r].q[1] = *(const u32x4*)(pa + 8);
        }

        // Stage next A chunk into the other buffer (its readers finished at
        // the barrier that ended iteration i-1).
        *(u32x4*)&ldsA[(i + 1) & 1][lA_off]     = an0;
        *(u32x4*)&ldsA[(i + 1) & 1][lA_off + 8] = an1;

#pragma unroll
        for (int r = 0; r < 4; ++r)
#pragma unroll
            for (int c = 0; c < 2; ++c)
                acc[r][c] = __builtin_amdgcn_wmma_f32_16x16x32_bf16(
                    false, a[r].v, false, b[c].v, (short)0, acc[r][c],
                    false, false);

        __syncthreads();
    }

    // Peeled final chunk.
    {
        const int kc = (nk - 1) << 5;
        b[0].q[0] = *(const u32x4*)(Bfrag0 + kc);
        b[0].q[1] = *(const u32x4*)(Bfrag0 + kc + 8);
        b[1].q[0] = *(const u32x4*)(Bfrag1 + kc);
        b[1].q[1] = *(const u32x4*)(Bfrag1 + kc + 8);
        const bf16_t* lA = &ldsA[(nk - 1) & 1][0];
#pragma unroll
        for (int r = 0; r < 4; ++r) {
            const bf16_t* pa = &lA[(r * 16 + lr) * 40 + hi * 16];
            a[r].q[0] = *(const u32x4*)(pa);
            a[r].q[1] = *(const u32x4*)(pa + 8);
        }
#pragma unroll
        for (int r = 0; r < 4; ++r)
#pragma unroll
            for (int c = 0; c < 2; ++c)
                acc[r][c] = __builtin_amdgcn_wmma_f32_16x16x32_bf16(
                    false, a[r].v, false, b[c].v, (short)0, acc[r][c],
                    false, false);
    }

    // ---- epilogue: D layout = VGPR i -> row i + 8*hi, col = lr ----
#pragma unroll
    for (int r = 0; r < 4; ++r) {
#pragma unroll
        for (int c = 0; c < 2; ++c) {
#pragma unroll
            for (int i = 0; i < 8; ++i) {
                const int row = m0 + r * 16 + i + hi * 8;
                const int col = n0 + wn + c * 16 + lr;
                float v = acc[r][c][i];
                if (EPI == 0) {
                    if (bias) v += bias[col];
                    Cf[(size_t)row * N + col] = v;
                } else if (EPI == 1) {
                    if (bias) v += bias[col];
                    Cb[(size_t)row * N + col] = (bf16_t)v;
                } else if (EPI == 2) {
                    v += (float)Xt[(size_t)row * N + col];
                    const float s = sigmoidf_(v);
                    if (col < HID) {                  // z gate
                        zbuf[row * HID + col] = s;
                    } else {                          // r gate -> h*r operand
                        const int cc = col - HID;
                        aux[row * HID + cc] = (bf16_t)(hbuf[row * HID + cc] * s);
                    }
                } else {                              // EPI == 3
                    v += (float)Xt[(size_t)row * N + col];
                    const float ht = tanhf_(v);
                    const float zz = zbuf[row * N + col];
                    const float hp = hbuf[row * N + col];
                    const float hn = (1.0f - zz) * hp + zz * ht;
                    hbuf[row * N + col] = hn;
                    const bf16_t hb = (bf16_t)hn;
                    aux[row * N + col]  = hb;         // next-step bf16 h
                    aux2[row * N + col] = hb;         // H[t] for final GEMM
                }
            }
        }
    }
}

// ---------------------------------------------------------------------------
// One-shot pack/convert. All weights are stored TRANSPOSED: Bt[n*K + k].
// Reads are coalesced; the strided writes are a one-time amortized cost.
// ---------------------------------------------------------------------------
__global__ void gru_pack_kernel(
    const float* __restrict__ x,    const float* __restrict__ h0,
    const float* __restrict__ W_zh, const float* __restrict__ W_zx, const float* __restrict__ b_z,
    const float* __restrict__ W_rh, const float* __restrict__ W_rx, const float* __restrict__ b_r,
    const float* __restrict__ W_hh, const float* __restrict__ W_hx,
    const float* __restrict__ W_out,
    bf16_t* __restrict__ W_zr_t, bf16_t* __restrict__ W_zrx_t,
    bf16_t* __restrict__ W_hh_t, bf16_t* __restrict__ W_hx_t,
    bf16_t* __restrict__ W_out_t,
    float* __restrict__ b_zr, bf16_t* __restrict__ x_b,
    float* __restrict__ h_f, bf16_t* __restrict__ h_b)
{
    const int S0 = HID * 2 * HID;            // W_zr  (K=HID,    N=2*HID)
    const int S1 = IN_DIM * 2 * HID;         // W_zrx (K=IN_DIM, N=2*HID)
    const int S2 = HID * HID;                // W_hh
    const int S3 = IN_DIM * HID;             // W_hx
    const int S4 = HID * OUT_DIM;            // W_out (K=HID,    N=OUT_DIM)
    const int S5 = 2 * HID;                  // b_zr
    const int S6 = T_STEPS * BATCH * IN_DIM; // x
    const int S7 = BATCH * HID;              // h0
    const int total = S0 + S1 + S2 + S3 + S4 + S5 + S6 + S7;

    for (int idx = blockIdx.x * blockDim.x + threadIdx.x; idx < total;
         idx += gridDim.x * blockDim.x) {
        int i = idx;
        if (i < S0) {   // i = k*(2*HID) + n over W_zh|W_rh; write [n][k]
            const int k = i / (2 * HID), n = i % (2 * HID);
            const float v = (n < HID) ? W_zh[k * HID + n] : W_rh[k * HID + (n - HID)];
            W_zr_t[(size_t)n * HID + k] = (bf16_t)v;  continue;
        }
        i -= S0;
        if (i < S1) {
            const int k = i / (2 * HID), n = i % (2 * HID);
            const float v = (n < HID) ? W_zx[k * HID + n] : W_rx[k * HID + (n - HID)];
            W_zrx_t[(size_t)n * IN_DIM + k] = (bf16_t)v; continue;
        }
        i -= S1;
        if (i < S2) {
            const int k = i / HID, n = i % HID;
            W_hh_t[(size_t)n * HID + k] = (bf16_t)W_hh[i];  continue;
        }
        i -= S2;
        if (i < S3) {
            const int k = i / HID, n = i % HID;
            W_hx_t[(size_t)n * IN_DIM + k] = (bf16_t)W_hx[i];  continue;
        }
        i -= S3;
        if (i < S4) {
            const int k = i / OUT_DIM, n = i % OUT_DIM;
            W_out_t[(size_t)n * HID + k] = (bf16_t)W_out[i]; continue;
        }
        i -= S4;
        if (i < S5) { b_zr[i] = (i < HID) ? b_z[i] : b_r[i - HID]; continue; }
        i -= S5;
        if (i < S6) { x_b[i] = (bf16_t)x[i]; continue; }
        i -= S6;
        h_f[i] = h0[i];
        h_b[i] = (bf16_t)h0[i];
    }
}

__global__ void copy_hfinal_kernel(const float* __restrict__ h_f,
                                   float* __restrict__ out)
{
    const int i = blockIdx.x * blockDim.x + threadIdx.x;
    if (i < BATCH * HID) out[i] = h_f[i];
}

// ---------------------------------------------------------------------------
extern "C" void kernel_launch(void* const* d_in, const int* in_sizes, int n_in,
                              void* d_out, int out_size, void* d_ws, size_t ws_size,
                              hipStream_t stream)
{
    (void)in_sizes; (void)n_in; (void)out_size; (void)ws_size;
    const float* x     = (const float*)d_in[0];
    const float* h0    = (const float*)d_in[1];
    const float* W_zh  = (const float*)d_in[2];
    const float* W_zx  = (const float*)d_in[3];
    const float* b_z   = (const float*)d_in[4];
    const float* W_rh  = (const float*)d_in[5];
    const float* W_rx  = (const float*)d_in[6];
    const float* b_r   = (const float*)d_in[7];
    const float* W_hh  = (const float*)d_in[8];
    const float* W_hx  = (const float*)d_in[9];
    const float* b_h   = (const float*)d_in[10];
    const float* W_out = (const float*)d_in[11];
    const float* b_out = (const float*)d_in[12];
    float* out = (float*)d_out;

    uint8_t* ws = (uint8_t*)d_ws;
    size_t off = 0;
    auto alloc = [&](size_t bytes) -> uint8_t* {
        uint8_t* p = ws + off;
        off += (bytes + 255) & ~(size_t)255;
        return p;
    };
    bf16_t* W_zr_t  = (bf16_t*)alloc((size_t)HID * 2 * HID * 2);
    bf16_t* W_zrx_t = (bf16_t*)alloc((size_t)IN_DIM * 2 * HID * 2);
    bf16_t* W_hh_t  = (bf16_t*)alloc((size_t)HID * HID * 2);
    bf16_t* W_hx_t  = (bf16_t*)alloc((size_t)IN_DIM * HID * 2);
    bf16_t* W_out_t = (bf16_t*)alloc((size_t)HID * OUT_DIM * 2);
    float*  b_zr    = (float*)alloc((size_t)2 * HID * 4);
    bf16_t* x_b     = (bf16_t*)alloc((size_t)T_STEPS * BATCH * IN_DIM * 2);
    bf16_t* Xzr     = (bf16_t*)alloc((size_t)T_STEPS * BATCH * 2 * HID * 2);
    bf16_t* Xh      = (bf16_t*)alloc((size_t)T_STEPS * BATCH * HID * 2);
    bf16_t* Hall    = (bf16_t*)alloc((size_t)T_STEPS * BATCH * HID * 2);
    float*  h_f     = (float*)alloc((size_t)BATCH * HID * 4);
    bf16_t* h_b     = (bf16_t*)alloc((size_t)BATCH * HID * 2);
    float*  zbuf    = (float*)alloc((size_t)BATCH * HID * 4);
    bf16_t* hr_b    = (bf16_t*)alloc((size_t)BATCH * HID * 2);

    // Phase 0: pack/convert + state init (re-run each replay => deterministic)
    gru_pack_kernel<<<2048, 256, 0, stream>>>(
        x, h0, W_zh, W_zx, b_z, W_rh, W_rx, b_r, W_hh, W_hx, W_out,
        W_zr_t, W_zrx_t, W_hh_t, W_hx_t, W_out_t, b_zr, x_b, h_f, h_b);

    const int MB = T_STEPS * BATCH;   // 32768

    // Phase 1: x-dependent GEMMs hoisted out of the recurrence (huge M)
    gemm_bf16_wmma<1><<<dim3(2 * HID / 128, MB / 64), 128, 0, stream>>>(
        x_b, W_zrx_t, b_zr, nullptr, Xzr,
        nullptr, nullptr, nullptr, nullptr, nullptr, MB, 2 * HID, IN_DIM);
    gemm_bf16_wmma<1><<<dim3(HID / 128, MB / 64), 128, 0, stream>>>(
        x_b, W_hx_t, b_h, nullptr, Xh,
        nullptr, nullptr, nullptr, nullptr, nullptr, MB, HID, IN_DIM);

    // Phase 2: sequential recurrence, 2 fused GEMM launches per timestep
    for (int t = 0; t < T_STEPS; ++t) {
        const bf16_t* Xzr_t = Xzr + (size_t)t * BATCH * 2 * HID;
        const bf16_t* Xh_t  = Xh  + (size_t)t * BATCH * HID;
        bf16_t*       H_t   = Hall + (size_t)t * BATCH * HID;
        // z,r gates: h @ [W_zh|W_rh]; epilogue: sigmoid, z->f32, hr=bf16(h*r)
        gemm_bf16_wmma<2><<<dim3(2 * HID / 128, 1), 128, 0, stream>>>(
            h_b, W_zr_t, nullptr, nullptr, nullptr,
            Xzr_t, zbuf, h_f, hr_b, nullptr, BATCH, 2 * HID, HID);
        // candidate + state update: (h*r) @ W_hh; epilogue: tanh + blend
        gemm_bf16_wmma<3><<<dim3(HID / 128, 1), 128, 0, stream>>>(
            hr_b, W_hh_t, nullptr, nullptr, nullptr,
            Xh_t, zbuf, h_f, h_b, H_t, BATCH, HID, HID);
    }

    // Phase 3: output projection hoisted out of the loop
    gemm_bf16_wmma<0><<<dim3(OUT_DIM / 128, MB / 64), 128, 0, stream>>>(
        Hall, W_out_t, b_out, out, nullptr,
        nullptr, nullptr, nullptr, nullptr, nullptr, MB, OUT_DIM, HID);

    copy_hfinal_kernel<<<(BATCH * HID + 255) / 256, 256, 0, stream>>>(
        h_f, out + (size_t)MB * OUT_DIM);
}
